// MPS_83434034692780
// MI455X (gfx1250) — matrix operand order repaired and checked by reading, not compiled, not run on hardware
//
#include <hip/hip_runtime.h>
#include <hip/hip_bf16.h>

typedef _Float16 v16h __attribute__((ext_vector_type(16)));
typedef _Float16 v8h  __attribute__((ext_vector_type(8)));
typedef float    v8f  __attribute__((ext_vector_type(8)));

#define BATCH 1024
#define SEQ   64
#define EDIM  256
#define NHEAD 4
#define HDIM  64
#define FFDIM 1024
#define MROWS (BATCH * SEQ)   // 65536

// ---------------------------------------------------------------------------
// fp32 -> f16 conversion (weights)
// ---------------------------------------------------------------------------
__global__ void f2h_kernel(const float* __restrict__ in, _Float16* __restrict__ out, int n) {
    int i = blockIdx.x * blockDim.x + threadIdx.x;
    if (i < n) out[i] = (_Float16)in[i];
}

// ---------------------------------------------------------------------------
// embedding gather: x[m,:] = table[cfg[m], :]; also f16 copy
// ---------------------------------------------------------------------------
__global__ __launch_bounds__(256) void embed_kernel(const int* __restrict__ cfg,
                                                    const float* __restrict__ table,
                                                    float* __restrict__ X,
                                                    _Float16* __restrict__ Xh) {
    int m = blockIdx.x;
    int e = threadIdx.x;
    int c = cfg[m];
    float v = table[c * EDIM + e];
    size_t idx = (size_t)m * EDIM + e;
    X[idx]  = v;
    Xh[idx] = (_Float16)v;
}

// ---------------------------------------------------------------------------
// Generic WMMA GEMM: C[m,n] = sum_k A[m,k] * W[n,k] + bias[n]
// A: (M,K) f16 row-major; W: (N,K) f16 row-major (i.e. x @ W^T).
// Each wave: 16x64 output tile; block = 8 waves = 128 rows.
// Emits v_wmma_f32_16x16x32_f16.
// ---------------------------------------------------------------------------
__global__ __launch_bounds__(256) void gemm_wmma(const _Float16* __restrict__ A,
                                                 const _Float16* __restrict__ W,
                                                 const float* __restrict__ bias,
                                                 void* __restrict__ Cout,
                                                 int N, int K, int out_half, int relu) {
    const int wave = threadIdx.x >> 5;   // 0..7
    const int lane = threadIdx.x & 31;
    const int lg   = lane >> 4;          // lane group (K-half select)
    const int ln16 = lane & 15;          // row (A) / col (B) within tile
    const int m0 = blockIdx.x * 128 + wave * 16;
    const int n0 = blockIdx.y * 64;

    v8f acc0 = {}, acc1 = {}, acc2 = {}, acc3 = {};

    const _Float16* arow = A + (size_t)(m0 + ln16) * K;
    const _Float16* wr0 = W + (size_t)(n0 +  0 + ln16) * K;
    const _Float16* wr1 = W + (size_t)(n0 + 16 + ln16) * K;
    const _Float16* wr2 = W + (size_t)(n0 + 32 + ln16) * K;
    const _Float16* wr3 = W + (size_t)(n0 + 48 + ln16) * K;

    for (int kt = 0; kt < K; kt += 32) {
        // A fragment: halves j: K = kt + lg*16 + j (contiguous 16, matches B)
        v16h a = *(const v16h*)(arow + kt + lg * 16);
        // B fragments: lane = column n, halves j: K = kt + lg*16 + j (contiguous 16)
        v16h b0 = *(const v16h*)(wr0 + kt + lg * 16);
        v16h b1 = *(const v16h*)(wr1 + kt + lg * 16);
        v16h b2 = *(const v16h*)(wr2 + kt + lg * 16);
        v16h b3 = *(const v16h*)(wr3 + kt + lg * 16);

        acc0 = __builtin_amdgcn_wmma_f32_16x16x32_f16(false, a, false, b0, (short)0, acc0, false, false);
        acc1 = __builtin_amdgcn_wmma_f32_16x16x32_f16(false, a, false, b1, (short)0, acc1, false, false);
        acc2 = __builtin_amdgcn_wmma_f32_16x16x32_f16(false, a, false, b2, (short)0, acc2, false, false);
        acc3 = __builtin_amdgcn_wmma_f32_16x16x32_f16(false, a, false, b3, (short)0, acc3, false, false);
    }

    // D layout: lane%16 = column, VGPR r = row (m0 + lg*8 + r)
    v8f accs[4] = {acc0, acc1, acc2, acc3};
#pragma unroll
    for (int t = 0; t < 4; ++t) {
        int col = n0 + t * 16 + ln16;
        float bv = bias ? bias[col] : 0.0f;
#pragma unroll
        for (int r = 0; r < 8; ++r) {
            int row = m0 + lg * 8 + r;
            float v = accs[t][r] + bv;
            if (relu) v = fmaxf(v, 0.0f);
            if (out_half) ((_Float16*)Cout)[(size_t)row * N + col] = (_Float16)v;
            else          ((float*)Cout)[(size_t)row * N + col]    = v;
        }
    }
}

// ---------------------------------------------------------------------------
// Attention: one block per (b,h); 64 threads = one query row each.
// qkv f16 layout: [m, 0..768) = q|k|v each 256 wide.
// ---------------------------------------------------------------------------
__global__ __launch_bounds__(64) void attn_kernel(const _Float16* __restrict__ qkv,
                                                  _Float16* __restrict__ Oh) {
    __shared__ _Float16 ks[SEQ][HDIM];
    __shared__ _Float16 vs[SEQ][HDIM];
    const int bh = blockIdx.x;
    const int b = bh >> 2;
    const int h = bh & 3;
    const int i = threadIdx.x;            // query index / row loader
    const size_t row  = (size_t)b * SEQ + i;
    const size_t base = row * (3 * EDIM);
    const int hoff = h * HDIM;

    // stage K,V rows into LDS (each thread loads its own l=i row)
#pragma unroll
    for (int d = 0; d < HDIM; d += 8) {
        *(v8h*)&ks[i][d] = *(const v8h*)(qkv + base + EDIM     + hoff + d);
        *(v8h*)&vs[i][d] = *(const v8h*)(qkv + base + 2 * EDIM + hoff + d);
    }
    float q[HDIM];
#pragma unroll
    for (int d = 0; d < HDIM; ++d) q[d] = (float)qkv[base + hoff + d];
    __syncthreads();

    const float scale = 0.125f;  // 1/sqrt(64)
    float s[SEQ];
    float mx = -3.0e30f;
    for (int j = 0; j < SEQ; ++j) {
        float acc = 0.0f;
#pragma unroll
        for (int d = 0; d < HDIM; ++d) acc += q[d] * (float)ks[j][d];
        acc *= scale;
        s[j] = acc;
        mx = fmaxf(mx, acc);
    }
    float sum = 0.0f;
    for (int j = 0; j < SEQ; ++j) { s[j] = __expf(s[j] - mx); sum += s[j]; }
    float inv = 1.0f / sum;

    float o[HDIM];
#pragma unroll
    for (int d = 0; d < HDIM; ++d) o[d] = 0.0f;
    for (int j = 0; j < SEQ; ++j) {
        float p = s[j] * inv;
#pragma unroll
        for (int d = 0; d < HDIM; ++d) o[d] += p * (float)vs[j][d];
    }
#pragma unroll
    for (int d = 0; d < HDIM; ++d)
        Oh[row * EDIM + hoff + d] = (_Float16)o[d];
}

// ---------------------------------------------------------------------------
// x = LayerNorm(x + delta); writes fp32 residual + f16 copy
// ---------------------------------------------------------------------------
__global__ __launch_bounds__(256) void add_ln_kernel(float* __restrict__ X,
                                                     const float* __restrict__ Delta,
                                                     const float* __restrict__ w,
                                                     const float* __restrict__ b,
                                                     _Float16* __restrict__ Xh) {
    __shared__ float r1[256];
    __shared__ float r2[256];
    const int m = blockIdx.x;
    const int e = threadIdx.x;
    const size_t idx = (size_t)m * EDIM + e;
    float v = X[idx] + Delta[idx];
    r1[e] = v;
    r2[e] = v * v;
    __syncthreads();
    for (int s = 128; s > 0; s >>= 1) {
        if (e < s) { r1[e] += r1[e + s]; r2[e] += r2[e + s]; }
        __syncthreads();
    }
    float mean = r1[0] * (1.0f / 256.0f);
    float var  = r2[0] * (1.0f / 256.0f) - mean * mean;
    float y = (v - mean) * rsqrtf(var + 1e-5f) * w[e] + b[e];
    X[idx]  = y;
    Xh[idx] = (_Float16)y;
}

// ---------------------------------------------------------------------------
// MPS chain: res = t0 @ t1 @ ... @ t63, output res[0,0] per batch.
// Only row 0 of t0 matters -> complex vector-matrix chain.
// 8 lanes per batch, 8 batches per 64-thread block.
// x[b,l] row layout: offset cfg*128 + i*16 + j*2 + {re,im}
// ---------------------------------------------------------------------------
__global__ __launch_bounds__(64) void mps_kernel(const float* __restrict__ X,
                                                 const int* __restrict__ cfg,
                                                 float* __restrict__ out) {
    __shared__ float2 vv[8][8];     // [local batch][j]
    const int lb = threadIdx.x >> 3;
    const int j  = threadIdx.x & 7;
    const int b  = blockIdx.x * 8 + lb;
    const size_t base0 = (size_t)b * SEQ * EDIM;

    int c0 = cfg[b * SEQ];
    const float* t0 = X + base0 + c0 * 128;     // row i=0 of first tensor
    vv[lb][j] = make_float2(t0[j * 2], t0[j * 2 + 1]);
    __syncthreads();

    for (int l = 1; l < SEQ; ++l) {
        int c = cfg[b * SEQ + l];
        const float* t = X + base0 + (size_t)l * EDIM + c * 128;
        float re = 0.0f, im = 0.0f;
#pragma unroll
        for (int k = 0; k < 8; ++k) {
            float2 a = vv[lb][k];
            float tr = t[k * 16 + j * 2];
            float ti = t[k * 16 + j * 2 + 1];
            re += a.x * tr - a.y * ti;
            im += a.x * ti + a.y * tr;
        }
        __syncthreads();
        vv[lb][j] = make_float2(re, im);
        __syncthreads();
    }
    if (j == 0) {
        out[2 * b]     = vv[lb][0].x;
        out[2 * b + 1] = vv[lb][0].y;
    }
}

// ---------------------------------------------------------------------------
extern "C" void kernel_launch(void* const* d_in, const int* in_sizes, int n_in,
                              void* d_out, int out_size, void* d_ws, size_t ws_size,
                              hipStream_t stream) {
    const int*   cfg   = (const int*)d_in[0];
    const float* table = (const float*)d_in[1];
    const float* in_w  = (const float*)d_in[2];
    const float* in_b  = (const float*)d_in[3];
    const float* ow    = (const float*)d_in[4];
    const float* ob    = (const float*)d_in[5];
    const float* f1w   = (const float*)d_in[6];
    const float* f1b   = (const float*)d_in[7];
    const float* f2w   = (const float*)d_in[8];
    const float* f2b   = (const float*)d_in[9];
    const float* ln1w  = (const float*)d_in[10];
    const float* ln1b  = (const float*)d_in[11];
    const float* ln2w  = (const float*)d_in[12];
    const float* ln2b  = (const float*)d_in[13];
    float* outp = (float*)d_out;
    (void)in_sizes; (void)n_in; (void)out_size; (void)ws_size;

    // -------- workspace sub-allocation --------
    char* ws = (char*)d_ws;
    size_t off = 0;
    auto alloc = [&](size_t bytes) -> void* {
        void* p = ws + off;
        off += (bytes + 255) & ~(size_t)255;
        return p;
    };
    const size_t M = MROWS;
    float*     X    = (float*)alloc(M * EDIM * 4);          //  64 MB residual
    _Float16*  Xh   = (_Float16*)alloc(M * EDIM * 2);       //  32 MB
    void*      REG  = alloc(M * 3 * EDIM * 2);              //  96 MB: qkv f16 / proj f32 / ff2out f32
    _Float16*  QKVh = (_Float16*)REG;
    float*     PROJ = (float*)REG;
    _Float16*  Oh   = (_Float16*)alloc(M * EDIM * 2);       //  32 MB
    _Float16*  Hh   = (_Float16*)alloc(M * FFDIM * 2);      // 128 MB
    _Float16*  w_in_h  = (_Float16*)alloc((size_t)2 * 768 * 256 * 2);
    _Float16*  w_out_h = (_Float16*)alloc((size_t)2 * 256 * 256 * 2);
    _Float16*  w_f1_h  = (_Float16*)alloc((size_t)2 * 1024 * 256 * 2);
    _Float16*  w_f2_h  = (_Float16*)alloc((size_t)2 * 256 * 1024 * 2);

    auto cvt = [&](const float* src, _Float16* dst, int n) {
        f2h_kernel<<<(n + 255) / 256, 256, 0, stream>>>(src, dst, n);
    };
    cvt(in_w, w_in_h,  2 * 768 * 256);
    cvt(ow,   w_out_h, 2 * 256 * 256);
    cvt(f1w,  w_f1_h,  2 * 1024 * 256);
    cvt(f2w,  w_f2_h,  2 * 256 * 1024);

    embed_kernel<<<(unsigned)M, 256, 0, stream>>>(cfg, table, X, Xh);

    const unsigned mg = (unsigned)(M / 128);
    for (int i = 0; i < 2; ++i) {
        // qkv = x @ Win^T + b  -> f16
        gemm_wmma<<<dim3(mg, 768 / 64), 256, 0, stream>>>(
            Xh, w_in_h + (size_t)i * 768 * 256, in_b + (size_t)i * 768,
            QKVh, 768, 256, /*out_half=*/1, /*relu=*/0);
        // attention -> f16 o
        attn_kernel<<<BATCH * NHEAD, 64, 0, stream>>>(QKVh, Oh);
        // proj = o @ Wout^T + b -> f32 (reuses qkv region)
        gemm_wmma<<<dim3(mg, 256 / 64), 256, 0, stream>>>(
            Oh, w_out_h + (size_t)i * 256 * 256, ob + (size_t)i * 256,
            PROJ, 256, 256, 0, 0);
        // x = LN(x + proj)
        add_ln_kernel<<<(unsigned)M, 256, 0, stream>>>(
            X, PROJ, ln1w + (size_t)i * 256, ln1b + (size_t)i * 256, Xh);
        // h = relu(x @ W1^T + b) -> f16
        gemm_wmma<<<dim3(mg, 1024 / 64), 256, 0, stream>>>(
            Xh, w_f1_h + (size_t)i * 1024 * 256, f1b + (size_t)i * 1024,
            Hh, 1024, 256, 1, 1);
        // f = h @ W2^T + b -> f32 (reuses qkv region)
        gemm_wmma<<<dim3(mg, 256 / 64), 256, 0, stream>>>(
            Hh, w_f2_h + (size_t)i * 256 * 1024, f2b + (size_t)i * 256,
            PROJ, 256, 1024, 0, 0);
        // x = LN(x + f)
        add_ln_kernel<<<(unsigned)M, 256, 0, stream>>>(
            X, PROJ, ln2w + (size_t)i * 256, ln2b + (size_t)i * 256, Xh);
    }

    mps_kernel<<<BATCH / 8, 64, 0, stream>>>(X, cfg, outp);
}
